// CausalSelfAttention_81879256531229
// MI455X (gfx1250) — compile-verified, matrix operand order checked by reading, not run on hardware
//
#include <hip/hip_runtime.h>
#include <hip/hip_bf16.h>

// ---------------------------------------------------------------------------
// CausalSelfAttention for MI455X (gfx1250): bf16 WMMA pipeline.
//   B=4, T=2048, C=1024, H=16, D=64
// Pipeline: cast(x,Wa,Wp)->bf16 ; QKV GEMM (WMMA, TDM-staged A tiles) ->
//           q/k/v [B,H,T,D]bf16 ; flash attention (WMMA, online softmax) ->
//           y [B,T,C]bf16 ; output projection GEMM (WMMA) -> f32 out.
// ---------------------------------------------------------------------------

typedef __attribute__((ext_vector_type(16))) __bf16        v16bf;
typedef __attribute__((ext_vector_type(8)))  float         v8f;
typedef __attribute__((ext_vector_type(4)))  unsigned int  v4u;
typedef __attribute__((ext_vector_type(8)))  int           v8i;
typedef __attribute__((ext_vector_type(4)))  int           v4i;

union FragU { uint4 q[2]; v16bf v; };

#if __has_builtin(__builtin_amdgcn_tensor_load_to_lds)
#define USE_TDM 1
#endif

// Load a 16x32 bf16 A/B^T fragment from a row-major (stride in elements) tile.
// Per ISA layout: lanes 0-15 hold row (lane&15), K chunks {0..7,16..23};
// lanes 16-31 hold row (lane&15), K chunks {8..15,24..31}. Both chunks are
// contiguous 16B -> two b128 loads.
__device__ __forceinline__ v16bf load_frag(const unsigned short* base, int stride, int lane) {
  const unsigned short* p = base + (size_t)(lane & 15) * stride + ((lane >> 4) & 1) * 8;
  FragU f;
  f.q[0] = *(const uint4*)(p);
  f.q[1] = *(const uint4*)(p + 16);
  return f.v;
}

__device__ __forceinline__ v8f bf16_wmma(v16bf a, v16bf b, v8f c) {
  return __builtin_amdgcn_wmma_f32_16x16x32_bf16(false, a, false, b, (short)0, c, false, false);
}

__device__ __forceinline__ unsigned short f2bf(float x) {
  unsigned int u = __float_as_uint(x);
  u += 0x7FFFu + ((u >> 16) & 1u);   // round-to-nearest-even
  return (unsigned short)(u >> 16);
}

#ifdef USE_TDM
// TDM: async 2-D tile load (tile_h rows x tile_w elems, 2-byte elements) from
// a row-major tensor (tensor_w x tensor_h, row stride = stride_elems) into
// LDS at byte offset lds_off, packed row-major [tile_h][tile_w].
// Descriptor layout per CDNA5 ISA.
__device__ __forceinline__ void tdm_load_tile_2d(unsigned lds_off, const void* gaddr,
                                                 unsigned tensor_w, unsigned tensor_h,
                                                 unsigned tile_w, unsigned tile_h,
                                                 unsigned long long stride_elems) {
  unsigned long long ga = (unsigned long long)(size_t)gaddr;
  v4u g0 = { 1u,                                   // count=1, flags=0
             lds_off,                              // lds_addr (bytes)
             (unsigned)ga,                         // global_addr[31:0]
             ((unsigned)(ga >> 32) & 0x1FFFFFFu) | (2u << 30) };  // addr[56:32] | type=2
  v8i g1 = { (int)(1u << 16),                      // wg_mask=0, data_size=1 (2 bytes)
             (int)(tensor_w << 16),                // dim0[15:0] at bits 63:48
             (int)((tensor_w >> 16) | (tensor_h << 16)),
             (int)((tensor_h >> 16) | (tile_w << 16)),
             (int)tile_h,                          // tile_dim1 (tile_dim2=0)
             (int)(unsigned)(stride_elems & 0xFFFFFFFFull),
             (int)(unsigned)(stride_elems >> 32),  // stride[47:32]; dim1_stride=0
             0 };
  v4i z = {0, 0, 0, 0};
#if __clang_major__ >= 23
  v8i z8 = {0, 0, 0, 0, 0, 0, 0, 0};
  __builtin_amdgcn_tensor_load_to_lds(g0, g1, z, z, z8, 0);
#else
  __builtin_amdgcn_tensor_load_to_lds(g0, g1, z, z, 0);
#endif
}
#endif

// ---------------------------------------------------------------------------
// Elementwise f32 -> bf16 cast (vectorized: 4 floats -> 8 bytes per thread).
// ---------------------------------------------------------------------------
__global__ __launch_bounds__(256) void cast_bf16_kernel(const float* __restrict__ s,
                                                        unsigned short* __restrict__ d,
                                                        int n4) {
  int i = blockIdx.x * blockDim.x + threadIdx.x;
  if (i < n4) {
    float4 f = ((const float4*)s)[i];
    unsigned int lo = (unsigned int)f2bf(f.x) | ((unsigned int)f2bf(f.y) << 16);
    unsigned int hi = (unsigned int)f2bf(f.z) | ((unsigned int)f2bf(f.w) << 16);
    ((uint2*)d)[i] = make_uint2(lo, hi);
  }
}

// ---------------------------------------------------------------------------
// Shared GEMM tile body: block tile 128x64, 8 waves (4x2), wave tile 32x32,
// K-step 64 (8 WMMAs per barrier pair). A tile staged by TDM when available.
// ---------------------------------------------------------------------------
#define GEMM_BODY(APTR, BPTR, KDIM, NDIM)                                         \
  const int tid  = threadIdx.x;                                                   \
  const int lane = tid & 31;                                                      \
  const int w    = tid >> 5;                                                      \
  const int wm   = w & 3;                                                         \
  const int wn   = w >> 2;                                                        \
  const int m0   = blockIdx.y * 128;                                              \
  const int n0   = blockIdx.x * 64;                                               \
  v8f acc[2][2] = {};                                                             \
  for (int k0 = 0; k0 < (KDIM); k0 += 64) {                                       \
    STAGE_A(APTR, KDIM)                                                           \
    for (int u = tid; u < 4096; u += 256) {                                       \
      int kk = u >> 6;                                                            \
      int n  = u & 63;                                                            \
      ldsB[n * 64 + kk] = (BPTR)[(size_t)(k0 + kk) * (NDIM) + n0 + n];            \
    }                                                                             \
    __syncthreads();                                                              \
    v16bf af[2][2], bfr[2][2];                                                    \
    _Pragma("unroll")                                                             \
    for (int f = 0; f < 2; ++f)                                                   \
      _Pragma("unroll")                                                           \
      for (int kc = 0; kc < 2; ++kc) {                                            \
        af[f][kc]  = load_frag(&ldsA[(wm * 32 + f * 16) * 64 + kc * 32], 64, lane);\
        bfr[f][kc] = load_frag(&ldsB[(wn * 32 + f * 16) * 64 + kc * 32], 64, lane);\
      }                                                                           \
    _Pragma("unroll")                                                             \
    for (int fm = 0; fm < 2; ++fm)                                                \
      _Pragma("unroll")                                                           \
      for (int fn = 0; fn < 2; ++fn) {                                            \
        acc[fm][fn] = bf16_wmma(af[fm][0], bfr[fn][0], acc[fm][fn]);              \
        acc[fm][fn] = bf16_wmma(af[fm][1], bfr[fn][1], acc[fm][fn]);              \
      }                                                                           \
    __syncthreads();                                                              \
  }

#ifdef USE_TDM
#define STAGE_A(APTR, KDIM)                                                       \
    if (tid == 0) {                                                               \
      tdm_load_tile_2d((unsigned)(size_t)(void*)ldsA,                             \
                       &(APTR)[(size_t)m0 * (KDIM) + k0],                         \
                       (KDIM), 8192u, 64u, 128u, (unsigned long long)(KDIM));     \
      __builtin_amdgcn_s_wait_tensorcnt(0);                                       \
    }
#else
#define STAGE_A(APTR, KDIM)                                                       \
    for (int u = tid; u < 1024; u += 256) {                                       \
      int row = u >> 3;                                                           \
      int c8  = (u & 7) << 3;                                                     \
      *(uint4*)&ldsA[row * 64 + c8] =                                             \
          *(const uint4*)&(APTR)[(size_t)(m0 + row) * (KDIM) + k0 + c8];          \
    }
#endif

// ---------------------------------------------------------------------------
// QKV GEMM: [8192,1024]bf16 x [1024,3072]bf16 + bias -> scatter to q/k/v
// [B,H,T,D] bf16.
// ---------------------------------------------------------------------------
__global__ __launch_bounds__(256) void qkv_gemm_kernel(
    const unsigned short* __restrict__ xb,   // [8192,1024]
    const unsigned short* __restrict__ wab,  // [1024,3072]
    const float* __restrict__ bias,          // [3072]
    unsigned short* __restrict__ qout,
    unsigned short* __restrict__ kout,
    unsigned short* __restrict__ vout) {
  constexpr int T = 2048, H = 16, Dh = 64;
  __shared__ __align__(16) unsigned short ldsA[128 * 64];
  __shared__ __align__(16) unsigned short ldsB[64 * 64];   // B^T: [n][k]

  GEMM_BODY(xb, wab, 1024, 3072)

  // epilogue: bias add + scatter to q/k/v in [B,H,T,D]
  const int half = lane >> 4;
  const int l15  = lane & 15;
#pragma unroll
  for (int fm = 0; fm < 2; ++fm)
#pragma unroll
    for (int fn = 0; fn < 2; ++fn)
#pragma unroll
      for (int i = 0; i < 8; ++i) {
        int m = m0 + wm * 32 + fm * 16 + i + half * 8;
        int n = n0 + wn * 32 + fn * 16 + l15;
        float val = acc[fm][fn][i] + bias[n];
        int which = n >> 10;       // 0=q 1=k 2=v
        int c = n & 1023;
        int h = c >> 6;
        int d = c & 63;
        int b = m >> 11;           // T = 2048
        int t = m & (T - 1);
        size_t off = (((size_t)(b * H + h)) * T + t) * Dh + d;
        unsigned short bv = f2bf(val);
        if (which == 0)       qout[off] = bv;
        else if (which == 1)  kout[off] = bv;
        else                  vout[off] = bv;
      }
}

// ---------------------------------------------------------------------------
// Output projection: [8192,1024]bf16 x [1024,1024]bf16 + bias -> f32 out.
// ---------------------------------------------------------------------------
__global__ __launch_bounds__(256) void proj_gemm_kernel(
    const unsigned short* __restrict__ yb,   // [8192,1024]
    const unsigned short* __restrict__ wpb,  // [1024,1024]
    const float* __restrict__ bias,          // [1024]
    float* __restrict__ out) {
  __shared__ __align__(16) unsigned short ldsA[128 * 64];
  __shared__ __align__(16) unsigned short ldsB[64 * 64];

  GEMM_BODY(yb, wpb, 1024, 1024)

  const int half = lane >> 4;
  const int l15  = lane & 15;
#pragma unroll
  for (int fm = 0; fm < 2; ++fm)
#pragma unroll
    for (int fn = 0; fn < 2; ++fn)
#pragma unroll
      for (int i = 0; i < 8; ++i) {
        int m = m0 + wm * 32 + fm * 16 + i + half * 8;
        int n = n0 + wn * 32 + fn * 16 + l15;
        out[(size_t)m * 1024 + n] = acc[fm][fn][i] + bias[n];
      }
}

// ---------------------------------------------------------------------------
// Flash attention: one wave per (32 q-rows, head). Online softmax, causal.
// ---------------------------------------------------------------------------
__global__ __launch_bounds__(32) void flash_attn_kernel(
    const unsigned short* __restrict__ qb,
    const unsigned short* __restrict__ kb,
    const unsigned short* __restrict__ vb,
    unsigned short* __restrict__ yb) {
  constexpr int T = 2048, Dh = 64, Cc = 1024;
  __shared__ __align__(16) unsigned short ldsP[2][16 * 32];
  __shared__ __align__(16) unsigned short ldsVt[64 * 32];   // V^T: [d][j]

  const int q0 = blockIdx.x * 32;
  const int bh = blockIdx.y;          // b*H + h
  const int bbatch = bh >> 4;
  const int hh     = bh & 15;
  const size_t head_off = (size_t)bh * T * Dh;
  const unsigned short* Q = qb + head_off;
  const unsigned short* K = kb + head_off;
  const unsigned short* V = vb + head_off;

  const int lane = threadIdx.x & 31;
  const int half = lane >> 4;
  const int l15  = lane & 15;

  // Q fragments: [mtile][kchunk], loaded once (global [T,D] row-major)
  v16bf qf[2][2];
#pragma unroll
  for (int mt = 0; mt < 2; ++mt)
#pragma unroll
    for (int kc = 0; kc < 2; ++kc)
      qf[mt][kc] = load_frag(Q + (size_t)(q0 + mt * 16) * Dh + kc * 32, Dh, lane);

  v8f O[2][4] = {};
  float rowm[2][8], rowl[2][8];
#pragma unroll
  for (int mt = 0; mt < 2; ++mt)
#pragma unroll
    for (int i = 0; i < 8; ++i) { rowm[mt][i] = -1e30f; rowl[mt][i] = 0.0f; }

  for (int j0 = 0; j0 <= q0; j0 += 32) {
    // stage V^T tile (32 keys x 64 dims -> 64 x 32)
    for (int idx = lane; idx < 64 * 32; idx += 32) {
      int d = idx & 63;
      int j = idx >> 6;
      ldsVt[d * 32 + j] = V[(size_t)(j0 + j) * Dh + d];
    }
    __syncthreads();

    // S = Q K^T  (K row-major [T,D] is exactly the B^T layout we need)
    v16bf kf[2][2];
#pragma unroll
    for (int nt = 0; nt < 2; ++nt)
#pragma unroll
      for (int kc = 0; kc < 2; ++kc)
        kf[nt][kc] = load_frag(K + (size_t)(j0 + nt * 16) * Dh + kc * 32, Dh, lane);

    v8f s[2][2];
#pragma unroll
    for (int mt = 0; mt < 2; ++mt)
#pragma unroll
      for (int nt = 0; nt < 2; ++nt) {
        v8f sv = {};
        sv = bf16_wmma(qf[mt][0], kf[nt][0], sv);
        sv = bf16_wmma(qf[mt][1], kf[nt][1], sv);
        s[mt][nt] = sv;
      }

    // scale + causal mask (C/D layout: row = i + 8*half, col = lane&15)
#pragma unroll
    for (int mt = 0; mt < 2; ++mt)
#pragma unroll
      for (int nt = 0; nt < 2; ++nt)
#pragma unroll
        for (int i = 0; i < 8; ++i) {
          int mrow = mt * 16 + i + half * 8;
          int ncol = nt * 16 + l15;
          float v = s[mt][nt][i] * 0.125f;   // 1/sqrt(64)
          s[mt][nt][i] = (j0 + ncol > q0 + mrow) ? -1e30f : v;
        }

    // online softmax update per row
#pragma unroll
    for (int mt = 0; mt < 2; ++mt) {
#pragma unroll
      for (int i = 0; i < 8; ++i) {
        float mi = fmaxf(s[mt][0][i], s[mt][1][i]);
#pragma unroll
        for (int msk = 8; msk >= 1; msk >>= 1)
          mi = fmaxf(mi, __shfl_xor(mi, msk, 32));
        float nm    = fmaxf(rowm[mt][i], mi);
        float alpha = __expf(rowm[mt][i] - nm);
        rowm[mt][i] = nm;
        float p0 = __expf(s[mt][0][i] - nm);
        float p1 = __expf(s[mt][1][i] - nm);
        ldsP[mt][(i + half * 8) * 32 + l15]      = f2bf(p0);
        ldsP[mt][(i + half * 8) * 32 + 16 + l15] = f2bf(p1);
        float ls = p0 + p1;
#pragma unroll
        for (int msk = 8; msk >= 1; msk >>= 1)
          ls += __shfl_xor(ls, msk, 32);
        rowl[mt][i] = rowl[mt][i] * alpha + ls;
#pragma unroll
        for (int dt = 0; dt < 4; ++dt)
          O[mt][dt][i] *= alpha;
      }
    }
    __syncthreads();

    // O += P V   (P re-layouted through LDS into A-fragment form)
    v16bf vf[4];
#pragma unroll
    for (int dt = 0; dt < 4; ++dt)
      vf[dt] = load_frag(&ldsVt[(dt * 16) * 32], 32, lane);
#pragma unroll
    for (int mt = 0; mt < 2; ++mt) {
      v16bf pa = load_frag(&ldsP[mt][0], 32, lane);
#pragma unroll
      for (int dt = 0; dt < 4; ++dt)
        O[mt][dt] = bf16_wmma(pa, vf[dt], O[mt][dt]);
    }
    __syncthreads();
  }

  // epilogue: normalize and store y [B,T,C] bf16
#pragma unroll
  for (int mt = 0; mt < 2; ++mt)
#pragma unroll
    for (int dt = 0; dt < 4; ++dt)
#pragma unroll
      for (int i = 0; i < 8; ++i) {
        int mrow = q0 + mt * 16 + i + half * 8;
        int d    = dt * 16 + l15;
        float o  = O[mt][dt][i] / rowl[mt][i];
        yb[((size_t)bbatch * T + mrow) * Cc + hh * Dh + d] = f2bf(o);
      }
}

// ---------------------------------------------------------------------------
extern "C" void kernel_launch(void* const* d_in, const int* in_sizes, int n_in,
                              void* d_out, int out_size, void* d_ws, size_t ws_size,
                              hipStream_t stream) {
  (void)in_sizes; (void)n_in; (void)out_size; (void)ws_size;

  const float* x      = (const float*)d_in[0];
  const float* W_attn = (const float*)d_in[1];
  const float* b_attn = (const float*)d_in[2];
  const float* W_proj = (const float*)d_in[3];
  const float* b_proj = (const float*)d_in[4];
  float* out = (float*)d_out;

  constexpr size_t NX  = 8192ull * 1024;   // x / q / k / v / y element count
  constexpr size_t NWA = 1024ull * 3072;
  constexpr size_t NWP = 1024ull * 1024;

  char* ws = (char*)d_ws;
  size_t off = 0;
  unsigned short* xb  = (unsigned short*)(ws + off); off += NX  * 2;
  unsigned short* wab = (unsigned short*)(ws + off); off += NWA * 2;
  unsigned short* wpb = (unsigned short*)(ws + off); off += NWP * 2;
  unsigned short* qb  = (unsigned short*)(ws + off); off += NX  * 2;
  unsigned short* kb  = (unsigned short*)(ws + off); off += NX  * 2;
  unsigned short* vb  = (unsigned short*)(ws + off); off += NX  * 2;
  unsigned short* yb  = (unsigned short*)(ws + off); off += NX  * 2;

  cast_bf16_kernel<<<(int)((NX / 4 + 255) / 256), 256, 0, stream>>>(x, xb, (int)(NX / 4));
  cast_bf16_kernel<<<(int)((NWA / 4 + 255) / 256), 256, 0, stream>>>(W_attn, wab, (int)(NWA / 4));
  cast_bf16_kernel<<<(int)((NWP / 4 + 255) / 256), 256, 0, stream>>>(W_proj, wpb, (int)(NWP / 4));

  qkv_gemm_kernel<<<dim3(48, 64), 256, 0, stream>>>(xb, wab, b_attn, qb, kb, vb);
  flash_attn_kernel<<<dim3(64, 64), 32, 0, stream>>>(qb, kb, vb, yb);
  proj_gemm_kernel<<<dim3(16, 64), 256, 0, stream>>>(yb, wpb, b_proj, out);
}
